// DSA_Equal_36009005809717
// MI455X (gfx1250) — matrix operand order repaired and checked by reading, not compiled, not run on hardware
//
#include <hip/hip_runtime.h>
#include <math.h>

typedef __bf16 bf16;
typedef __attribute__((ext_vector_type(16))) __bf16 v16bf;
typedef __attribute__((ext_vector_type(8)))  float  v8f;

// ---------------------------------------------------------------------------
// Shapes (fixed by the reference): B=2, C=I=64, H=W=64, N=4096
// ---------------------------------------------------------------------------
#define NB   2
#define CDIM 64
#define NPIX 4096

static __device__ __forceinline__ v8f wmma_bf16(v16bf a, v16bf b, v8f c) {
  // v_wmma_f32_16x16x32_bf16: D = A(16x32) * B(32x16) + C(16x16 f32)
  return __builtin_amdgcn_wmma_f32_16x16x32_bf16(false, a, false, b, (short)0, c,
                                                 false, false);
}

// A-fragment (16x32 bf16). Source: row-major [row][k] with `row` pointing at the
// lane's row. Element e (= g*8 + w): K = k0 + hi*16 + g*8 + w.
static __device__ __forceinline__ v16bf load_a_frag(const bf16* __restrict__ row,
                                                    int k0, int hi) {
  union { v16bf v; bf16 e[16]; } u;
#pragma unroll
  for (int g = 0; g < 2; ++g)
#pragma unroll
    for (int w = 0; w < 8; ++w)
      u.e[g * 8 + w] = row[k0 + hi * 16 + g * 8 + w];
  return u.v;
}

// B-fragment (32x16 bf16). Source: [col][k] row-major with `row` at the lane's
// column (col = lane&15). Element e: K = k0 + hi*16 + e  -> one contiguous 32B.
static __device__ __forceinline__ v16bf load_b_frag(const bf16* __restrict__ row,
                                                    int k0, int hi) {
  union { v16bf v; bf16 e[16]; } u;
#pragma unroll
  for (int e = 0; e < 16; ++e) u.e[e] = row[k0 + hi * 16 + e];
  return u.v;
}

// ---------------------------------------------------------------------------
// K0: f/g masking + 6 projections, outputs bf16 in WMMA-friendly layouts.
//   Qt/Kt : [B][N][I]  (n-major, 64 bf16 per row)
//   V     : [B][I][N]
// ---------------------------------------------------------------------------
__global__ void proj_kernel(const float* __restrict__ x, const float* __restrict__ mask,
                            const float* __restrict__ phi_s, const float* __restrict__ theta_s,
                            const float* __restrict__ gate_s,
                            const float* __restrict__ phi_c, const float* __restrict__ theta_c,
                            const float* __restrict__ gate_c,
                            bf16* __restrict__ QtS, bf16* __restrict__ KtS, bf16* __restrict__ VS,
                            bf16* __restrict__ QtC, bf16* __restrict__ KtC, bf16* __restrict__ VC) {
  __shared__ float fs[64][65];
  __shared__ float gs[64][65];
  const int b  = blockIdx.y;
  const int n0 = blockIdx.x * 64;
  const int tid = threadIdx.x;

  for (int idx = tid; idx < 64 * 64; idx += 256) {
    int c = idx >> 6, j = idx & 63;
    float xv = x[((size_t)(b * 64 + c)) * NPIX + n0 + j];
    float mv = mask[(size_t)b * NPIX + n0 + j];
    float fv = xv * mv;
    fs[c][j] = fv;
    gs[c][j] = xv - fv;
  }
  __syncthreads();

  const int i  = tid >> 2;
  const int nq = tid & 3;
  const float* Wm[6] = {phi_s, theta_s, gate_s, phi_c, theta_c, gate_c};
  bf16*        Dm[6] = {QtS,   KtS,     VS,     QtC,   KtC,     VC};

  for (int m = 0; m < 6; ++m) {
    const float(*src)[65] = (m >= 4) ? gs : fs;  // only theta_c/gate_c use g
    const float* W = Wm[m];
    float acc[16];
#pragma unroll
    for (int k = 0; k < 16; ++k) acc[k] = 0.f;
    for (int c = 0; c < 64; ++c) {
      float wv = W[i * 64 + c];
#pragma unroll
      for (int k = 0; k < 16; ++k) acc[k] += wv * src[c][nq * 16 + k];
    }
    bf16* dst = Dm[m];
    if (m == 2 || m == 5) {  // V: [I][N]
      size_t base = (size_t)b * 64 * NPIX + (size_t)i * NPIX + n0 + nq * 16;
#pragma unroll
      for (int k = 0; k < 16; ++k) dst[base + k] = (bf16)acc[k];
    } else {                 // Qt/Kt: [N][I]
      size_t base = (size_t)b * NPIX * 64;
#pragma unroll
      for (int k = 0; k < 16; ++k)
        dst[base + (size_t)(n0 + nq * 16 + k) * 64 + i] = (bf16)acc[k];
    }
  }
}

// ---------------------------------------------------------------------------
// K1: S[n][m] = sum_i Qt[n][i]*Kt[m][i]   (raw logits into d_out score region)
// One wave: 16 n-rows x 128 m-cols (8 tiles), A-frags reused across tiles.
// grid = (N/128, N/128, B), block = 256 (8 waves stacked in n).
// ---------------------------------------------------------------------------
__global__ void score_kernel(const bf16* __restrict__ Qt, const bf16* __restrict__ Kt,
                             float* __restrict__ S) {
  const int lane = threadIdx.x & 31;
  const int wv   = threadIdx.x >> 5;
  const int lr   = lane & 15;
  const int hi   = lane >> 4;
  const int b    = blockIdx.z;
  const int n0   = blockIdx.x * 128 + wv * 16;
  const int m0   = blockIdx.y * 128;

  const bf16* Qb = Qt + (size_t)b * NPIX * 64;
  const bf16* Kb = Kt + (size_t)b * NPIX * 64;
  float*      Sb = S  + (size_t)b * NPIX * NPIX;

  const bf16* qrow = Qb + (size_t)(n0 + lr) * 64;
  v16bf a0 = load_a_frag(qrow, 0, hi);
  v16bf a1 = load_a_frag(qrow, 32, hi);

#pragma unroll
  for (int mt = 0; mt < 8; ++mt) {
    const bf16* krow = Kb + (size_t)(m0 + mt * 16 + lr) * 64;
    v16bf b0 = load_b_frag(krow, 0, hi);
    v16bf b1 = load_b_frag(krow, 32, hi);
    v8f acc = {0.f, 0.f, 0.f, 0.f, 0.f, 0.f, 0.f, 0.f};
    acc = wmma_bf16(a0, b0, acc);
    acc = wmma_bf16(a1, b1, acc);
#pragma unroll
    for (int j = 0; j < 8; ++j)
      Sb[(size_t)(n0 + j + 8 * hi) * NPIX + m0 + mt * 16 + lr] = acc[j];
  }
}

// ---------------------------------------------------------------------------
// K2: in-place row softmax. grid = B*N rows, block = 256 (16 elems/thread).
// ---------------------------------------------------------------------------
__global__ void softmax_kernel(float* __restrict__ S) {
  __shared__ float red[256];
  const int tid = threadIdx.x;
  float* row = S + (size_t)blockIdx.x * NPIX;
  float v[16];
  float mx = -1e30f;
#pragma unroll
  for (int i = 0; i < 16; ++i) { v[i] = row[tid + i * 256]; mx = fmaxf(mx, v[i]); }
  red[tid] = mx; __syncthreads();
  for (int s = 128; s > 0; s >>= 1) {
    if (tid < s) red[tid] = fmaxf(red[tid], red[tid + s]);
    __syncthreads();
  }
  mx = red[0]; __syncthreads();
  float sum = 0.f;
#pragma unroll
  for (int i = 0; i < 16; ++i) { v[i] = __expf(v[i] - mx); sum += v[i]; }
  red[tid] = sum; __syncthreads();
  for (int s = 128; s > 0; s >>= 1) {
    if (tid < s) red[tid] += red[tid + s];
    __syncthreads();
  }
  float inv = 1.0f / red[0];
#pragma unroll
  for (int i = 0; i < 16; ++i) row[tid + i * 256] = v[i] * inv;
}

// ---------------------------------------------------------------------------
// K3: feat[i][n] = sum_m V[i][m] * score[n][m].  One wave owns one 16-wide
// n-tile and all 64 i rows (4 accumulators); score rows read fp32, converted
// to bf16 in-register; sequential stream prefetched 4KB ahead.
// grid = (N/(16*8), B), block = 256.
// ---------------------------------------------------------------------------
__global__ void feat_kernel(const bf16* __restrict__ V, const float* __restrict__ S,
                            float* __restrict__ feat) {
  const int lane = threadIdx.x & 31;
  const int wv   = threadIdx.x >> 5;
  const int lr   = lane & 15;
  const int hi   = lane >> 4;
  const int b    = blockIdx.y;
  const int n0   = (blockIdx.x * 8 + wv) * 16;

  const bf16*  Vb = V    + (size_t)b * 64 * NPIX;
  const float* Sb = S    + (size_t)b * NPIX * NPIX;
  float*       Fb = feat + (size_t)b * 64 * NPIX;

  v8f acc[4];
#pragma unroll
  for (int t = 0; t < 4; ++t) acc[t] = (v8f){0.f,0.f,0.f,0.f,0.f,0.f,0.f,0.f};

  const float* srow = Sb + (size_t)(n0 + lr) * NPIX + hi * 16;
  for (int m0 = 0; m0 < NPIX; m0 += 32) {
    const float* sp = srow + m0;
    // stream-prefetch the score row 4KB ahead (global_prefetch_b8)
    __builtin_prefetch(sp + 1024, 0, 1);
    union { v16bf v; bf16 e[16]; } ub;
#pragma unroll
    for (int e = 0; e < 16; ++e) ub.e[e] = (bf16)sp[e];
#pragma unroll
    for (int t = 0; t < 4; ++t) {
      const bf16* vrow = Vb + (size_t)(t * 16 + lr) * NPIX + m0;
      v16bf a = load_a_frag(vrow, 0, hi);
      acc[t] = wmma_bf16(a, ub.v, acc[t]);
    }
  }
#pragma unroll
  for (int t = 0; t < 4; ++t)
#pragma unroll
    for (int j = 0; j < 8; ++j)
      Fb[(size_t)(t * 16 + j + 8 * hi) * NPIX + n0 + lr] = acc[t][j];
}

// ---------------------------------------------------------------------------
// K4: y = relu(Wout @ [W_s@feat_s + x ; W_c@feat_c + x] + b); per-channel
// sum / sumsq accumulated via float atomics. grid = (N/64, B), block = 256.
// ---------------------------------------------------------------------------
__global__ void combine_kernel(const float* __restrict__ x,
                               const float* __restrict__ feat_s,
                               const float* __restrict__ feat_c,
                               const float* __restrict__ Wsp, const float* __restrict__ Wcp,
                               const float* __restrict__ Wout, const float* __restrict__ Wb,
                               float* __restrict__ y, float* __restrict__ sums) {
  __shared__ float ta[64][65];
  __shared__ float tb[64][65];
  const int b  = blockIdx.y;
  const int n0 = blockIdx.x * 64;
  const int tid = threadIdx.x;

  for (int idx = tid; idx < 64 * 64; idx += 256) {
    int c = idx >> 6, j = idx & 63;
    ta[c][j] = feat_s[((size_t)b * 64 + c) * NPIX + n0 + j];
    tb[c][j] = feat_c[((size_t)b * 64 + c) * NPIX + n0 + j];
  }
  __syncthreads();

  const int d  = tid >> 2;
  const int nq = tid & 3;
  float accS[16], accC[16];
#pragma unroll
  for (int k = 0; k < 16; ++k) { accS[k] = 0.f; accC[k] = 0.f; }
  for (int i = 0; i < 64; ++i) {
    float wsv = Wsp[d * 64 + i];
    float wcv = Wcp[d * 64 + i];
#pragma unroll
    for (int k = 0; k < 16; ++k) {
      accS[k] += wsv * ta[i][nq * 16 + k];
      accC[k] += wcv * tb[i][nq * 16 + k];
    }
  }
#pragma unroll
  for (int k = 0; k < 16; ++k) {
    float xv = x[((size_t)b * 64 + d) * NPIX + n0 + nq * 16 + k];
    accS[k] += xv;
    accC[k] += xv;
  }
  __syncthreads();
#pragma unroll
  for (int k = 0; k < 16; ++k) { ta[d][nq * 16 + k] = accS[k]; tb[d][nq * 16 + k] = accC[k]; }
  __syncthreads();

  float accY[16];
  float bias = Wb[d];
#pragma unroll
  for (int k = 0; k < 16; ++k) accY[k] = bias;
  for (int i = 0; i < 64; ++i) {
    float w0 = Wout[d * 128 + i];
    float w1 = Wout[d * 128 + 64 + i];
#pragma unroll
    for (int k = 0; k < 16; ++k)
      accY[k] += w0 * ta[i][nq * 16 + k] + w1 * tb[i][nq * 16 + k];
  }
#pragma unroll
  for (int k = 0; k < 16; ++k) accY[k] = fmaxf(accY[k], 0.f);
  __syncthreads();
#pragma unroll
  for (int k = 0; k < 16; ++k) {
    ta[d][nq * 16 + k] = accY[k];
    y[((size_t)b * 64 + d) * NPIX + n0 + nq * 16 + k] = accY[k];
  }
  __syncthreads();
  if (tid < 64) {
    float s = 0.f, s2 = 0.f;
    for (int j = 0; j < 64; ++j) { float v = ta[tid][j]; s += v; s2 += v * v; }
    atomicAdd(&sums[tid], s);
    atomicAdd(&sums[64 + tid], s2);
  }
}

// ---------------------------------------------------------------------------
// K5: training-mode BatchNorm finalize.
// ---------------------------------------------------------------------------
__global__ void bn_kernel(const float* __restrict__ y, const float* __restrict__ sums,
                          const float* __restrict__ gamma, const float* __restrict__ beta,
                          float* __restrict__ out) {
  int idx = blockIdx.x * 256 + threadIdx.x;
  int c = (idx >> 12) & 63;  // [b][c][4096]
  const float cnt = (float)(NB * NPIX);
  float mean = sums[c] / cnt;
  float var  = sums[64 + c] / cnt - mean * mean;
  out[idx] = gamma[c] * (y[idx] - mean) * rsqrtf(var + 1e-5f) + beta[c];
}

__global__ void zero_kernel(float* __restrict__ p) { p[threadIdx.x] = 0.f; }

// ---------------------------------------------------------------------------
extern "C" void kernel_launch(void* const* d_in, const int* in_sizes, int n_in,
                              void* d_out, int out_size, void* d_ws, size_t ws_size,
                              hipStream_t stream) {
  (void)in_sizes; (void)n_in; (void)out_size; (void)ws_size;
  const float* x       = (const float*)d_in[0];
  const float* mask    = (const float*)d_in[1];
  const float* theta_s = (const float*)d_in[2];
  const float* phi_s   = (const float*)d_in[3];
  const float* gate_s  = (const float*)d_in[4];
  const float* W_s     = (const float*)d_in[5];
  const float* theta_c = (const float*)d_in[6];
  const float* phi_c   = (const float*)d_in[7];
  const float* gate_c  = (const float*)d_in[8];
  const float* W_c     = (const float*)d_in[9];
  const float* Wout_w  = (const float*)d_in[10];
  const float* Wout_b  = (const float*)d_in[11];
  const float* bn_g    = (const float*)d_in[12];
  const float* bn_b    = (const float*)d_in[13];

  // Output layout: out [2,64,64,64] ++ score_self [2,4096,4096] ++ score_cross
  float* out     = (float*)d_out;
  float* score_s = out + (size_t)NB * CDIM * NPIX;           // +524288
  float* score_c = score_s + (size_t)NB * NPIX * NPIX;       // +33554432

  // Scratch layout
  char* ws = (char*)d_ws;
  size_t off = 0;
  const size_t PROJ = (size_t)NB * NPIX * 64 * sizeof(bf16);   // 1 MB each
  const size_t FEAT = (size_t)NB * 64 * NPIX * sizeof(float);  // 2 MB each
  bf16* QtS = (bf16*)(ws + off); off += PROJ;
  bf16* KtS = (bf16*)(ws + off); off += PROJ;
  bf16* QtC = (bf16*)(ws + off); off += PROJ;
  bf16* KtC = (bf16*)(ws + off); off += PROJ;
  bf16* VS  = (bf16*)(ws + off); off += PROJ;
  bf16* VC  = (bf16*)(ws + off); off += PROJ;
  float* featS = (float*)(ws + off); off += FEAT;
  float* featC = (float*)(ws + off); off += FEAT;
  float* yv    = (float*)(ws + off); off += FEAT;
  float* sums  = (float*)(ws + off); off += 256 * sizeof(float);

  zero_kernel<<<1, 128, 0, stream>>>(sums);

  proj_kernel<<<dim3(NPIX / 64, NB), 256, 0, stream>>>(
      x, mask, phi_s, theta_s, gate_s, phi_c, theta_c, gate_c,
      QtS, KtS, VS, QtC, KtC, VC);

  // Per-branch chains: each branch's 134MB score working set stays L2-resident
  // (192MB L2) across its producer->softmax->feat consumers.
  score_kernel<<<dim3(NPIX / 128, NPIX / 128, NB), 256, 0, stream>>>(QtS, KtS, score_s);
  softmax_kernel<<<NB * NPIX, 256, 0, stream>>>(score_s);
  feat_kernel<<<dim3(NPIX / 128, NB), 256, 0, stream>>>(VS, score_s, featS);

  score_kernel<<<dim3(NPIX / 128, NPIX / 128, NB), 256, 0, stream>>>(QtC, KtC, score_c);
  softmax_kernel<<<NB * NPIX, 256, 0, stream>>>(score_c);
  feat_kernel<<<dim3(NPIX / 128, NB), 256, 0, stream>>>(VC, score_c, featC);

  combine_kernel<<<dim3(NPIX / 64, NB), 256, 0, stream>>>(
      x, featS, featC, W_s, W_c, Wout_w, Wout_b, yv, sums);

  bn_kernel<<<(NB * CDIM * NPIX) / 256, 256, 0, stream>>>(yv, sums, bn_g, bn_b, out);
}
